// MoELayer_88218628260656
// MI455X (gfx1250) — matrix operand order repaired and checked by reading, not compiled, not run on hardware
//
#include <hip/hip_runtime.h>
#include <stdint.h>

// Problem sizes (match reference)
#define T_TOK 8192   // B*S
#define DIM   1024   // D
#define HID   4096   // H
#define NE    8      // E

// ---------------------------------------------------------------------------
// helpers
// ---------------------------------------------------------------------------
typedef __attribute__((ext_vector_type(16))) __bf16 v16bf;
typedef __attribute__((ext_vector_type(8)))  float  v8f;
typedef __attribute__((ext_vector_type(4)))  unsigned int uint4v;

__device__ __forceinline__ unsigned short f2bf(float f) {
    // round-to-nearest-even fp32 -> bf16 (bit pattern)
    unsigned int u = __float_as_uint(f);
    u += 0x7FFFu + ((u >> 16) & 1u);
    return (unsigned short)(u >> 16);
}

__device__ __forceinline__ v16bf load_frag(const unsigned short* p0,
                                           const unsigned short* p1) {
    v16bf f;
    ((uint4v*)&f)[0] = *(const uint4v*)p0;   // global_load_b128
    ((uint4v*)&f)[1] = *(const uint4v*)p1;   // global_load_b128
    return f;
}

// ---------------------------------------------------------------------------
// K0: zero per-expert counters
// ---------------------------------------------------------------------------
__global__ void k_zero_counts(int* counts) {
    if (threadIdx.x < NE) counts[threadIdx.x] = 0;
}

// ---------------------------------------------------------------------------
// K1: weight-norm, fp32 output (gate weights). One wave32 per row.
// ---------------------------------------------------------------------------
__global__ void k_rownorm_f32(const float* __restrict__ v,
                              const float* __restrict__ g,
                              float* __restrict__ outp,
                              int rows, int len) {
    int wave = (int)((blockIdx.x * blockDim.x + threadIdx.x) >> 5);
    int lane = threadIdx.x & 31;
    if (wave >= rows) return;
    const float4* src = (const float4*)(v + (long)wave * len);
    int n4 = len >> 2;
    float s = 0.f;
    for (int i = lane; i < n4; i += 32) {
        float4 q = src[i];
        s += q.x*q.x + q.y*q.y + q.z*q.z + q.w*q.w;
    }
    #pragma unroll
    for (int m = 16; m >= 1; m >>= 1) s += __shfl_xor(s, m, 32);
    float scale = g[wave] / fmaxf(sqrtf(s), 1e-12f);
    float4* dst = (float4*)(outp + (long)wave * len);
    for (int i = lane; i < n4; i += 32) {
        float4 q = src[i];
        q.x *= scale; q.y *= scale; q.z *= scale; q.w *= scale;
        dst[i] = q;
    }
}

// ---------------------------------------------------------------------------
// K2/K3: weight-norm, bf16 output (expert weights). One wave32 per row.
// ---------------------------------------------------------------------------
__global__ void k_rownorm_bf16(const float* __restrict__ v,
                               const float* __restrict__ g,
                               unsigned short* __restrict__ outp,
                               int rows, int len) {
    int wave = (int)((blockIdx.x * blockDim.x + threadIdx.x) >> 5);
    int lane = threadIdx.x & 31;
    if (wave >= rows) return;
    const float4* src = (const float4*)(v + (long)wave * len);
    int n4 = len >> 2;
    float s = 0.f;
    for (int i = lane; i < n4; i += 32) {
        float4 q = src[i];
        s += q.x*q.x + q.y*q.y + q.z*q.z + q.w*q.w;
    }
    #pragma unroll
    for (int m = 16; m >= 1; m >>= 1) s += __shfl_xor(s, m, 32);
    float scale = g[wave] / fmaxf(sqrtf(s), 1e-12f);
    uint2* dst = (uint2*)(outp + (long)wave * len);
    for (int i = lane; i < n4; i += 32) {
        float4 q = src[i];
        uint2 p;
        p.x = (unsigned)f2bf(q.x*scale) | ((unsigned)f2bf(q.y*scale) << 16);
        p.y = (unsigned)f2bf(q.z*scale) | ((unsigned)f2bf(q.w*scale) << 16);
        dst[i] = p;
    }
}

// ---------------------------------------------------------------------------
// K4: convert activations fp32 -> bf16. One wave32 per token row.
// ---------------------------------------------------------------------------
__global__ void k_cvt_x(const float* __restrict__ x,
                        unsigned short* __restrict__ xb) {
    int wave = (int)((blockIdx.x * blockDim.x + threadIdx.x) >> 5);
    int lane = threadIdx.x & 31;
    if (wave >= T_TOK) return;
    const float4* src = (const float4*)(x + (long)wave * DIM);
    uint2* dst = (uint2*)(xb + (long)wave * DIM);
    #pragma unroll 2
    for (int i = lane; i < DIM/4; i += 32) {
        float4 q = src[i];
        uint2 p;
        p.x = (unsigned)f2bf(q.x) | ((unsigned)f2bf(q.y) << 16);
        p.y = (unsigned)f2bf(q.z) | ((unsigned)f2bf(q.w) << 16);
        dst[i] = p;
    }
}

// ---------------------------------------------------------------------------
// K5: gating + top-2 routing. One wave32 per token. fp32 exact logits;
// softmax skipped (monotonic -> identical top-2).
// ---------------------------------------------------------------------------
__global__ void k_gate_route(const float* __restrict__ x,
                             const float* __restrict__ gwn,
                             const float* __restrict__ gate_b,
                             int* __restrict__ counts,
                             int* __restrict__ tok_idx,
                             int* __restrict__ tok_map) {
    int wave = (int)((blockIdx.x * blockDim.x + threadIdx.x) >> 5);
    int lane = threadIdx.x & 31;
    if (wave >= T_TOK) return;
    const float4* xr = (const float4*)(x + (long)wave * DIM);
    float acc[NE];
    #pragma unroll
    for (int e = 0; e < NE; e++) acc[e] = 0.f;
    for (int i = lane; i < DIM/4; i += 32) {
        float4 xv = xr[i];
        #pragma unroll
        for (int e = 0; e < NE; e++) {
            float4 w = ((const float4*)(gwn + (long)e * DIM))[i];
            acc[e] += xv.x*w.x + xv.y*w.y + xv.z*w.z + xv.w*w.w;
        }
    }
    #pragma unroll
    for (int e = 0; e < NE; e++) {
        #pragma unroll
        for (int m = 16; m >= 1; m >>= 1) acc[e] += __shfl_xor(acc[e], m, 32);
    }
    if (lane == 0) {
        float l[NE];
        #pragma unroll
        for (int e = 0; e < NE; e++) l[e] = acc[e] + gate_b[e];
        int b0 = 0;
        #pragma unroll
        for (int e = 1; e < NE; e++) if (l[e] > l[b0]) b0 = e;     // first max wins ties
        int b1 = (b0 == 0) ? 1 : 0;
        #pragma unroll
        for (int e = 0; e < NE; e++)
            if (e != b0 && l[e] > l[b1]) b1 = e;
        int s0 = atomicAdd(&counts[b0], 1);
        int s1 = atomicAdd(&counts[b1], 1);
        tok_idx[b0 * T_TOK + s0] = wave;
        tok_idx[b1 * T_TOK + s1] = wave;
        tok_map[wave*4 + 0] = b0; tok_map[wave*4 + 1] = s0;
        tok_map[wave*4 + 2] = b1; tok_map[wave*4 + 3] = s1;
    }
}

// ---------------------------------------------------------------------------
// K6: exclusive prefix over counts -> per-expert base into compacted buffers
// ---------------------------------------------------------------------------
__global__ void k_prefix(const int* __restrict__ counts, int* __restrict__ basep) {
    if (threadIdx.x == 0) {
        int b = 0;
        for (int e = 0; e < NE; e++) { basep[e] = b; b += counts[e]; }
    }
}

// ---------------------------------------------------------------------------
// K7: expert GEMM1 + Swish: h = swish(Xg @ W1^T + b1), bf16 out.
// Block = 8 waves, block tile 128(M) x 256(N); wave tile 64x64 =
// 4x4 v_wmma_f32_16x16x32_bf16 accumulators, K-loop over D in steps of 32.
// A rows gathered through tok_idx (per-expert routing).
// ---------------------------------------------------------------------------
__global__ __launch_bounds__(256)
void k_gemm1_swish(const unsigned short* __restrict__ xb,
                   const unsigned short* __restrict__ w1b,
                   const float* __restrict__ b1,
                   const int* __restrict__ counts,
                   const int* __restrict__ basep,
                   const int* __restrict__ tok_idx,
                   unsigned short* __restrict__ h) {
    const int e    = blockIdx.z;
    const int cnt  = counts[e];
    const int mblk = blockIdx.y * 128;
    if (mblk >= cnt) return;                 // sparse dispatch early-exit
    const int nblk = blockIdx.x * 256;
    const int wave = threadIdx.x >> 5;
    const int lane = threadIdx.x & 31;
    const int laneLo = lane & 15, laneHi = lane >> 4;
    const int wm = wave >> 2, wn = wave & 3;
    const int mW = mblk + wm * 64;
    const int nW = nblk + wn * 64;

    const int* tix = tok_idx + (long)e * T_TOK;
    long arow[4];
    #pragma unroll
    for (int i = 0; i < 4; i++) {
        int s = mW + i*16 + laneLo; if (s >= cnt) s = cnt - 1;
        arow[i] = (long)tix[s] * DIM;        // gather token row
    }
    const unsigned short* wbase = w1b + (long)e * HID * DIM;
    long brow[4];
    #pragma unroll
    for (int j = 0; j < 4; j++) brow[j] = (long)(nW + j*16 + laneLo) * DIM;

    const int aoff = laneHi ? 8 : 0;         // A: K 0-7 / 8-15 (+16)
    const int boff = laneHi ? 8 : 0;         // B: K 0-7 / 8-15 (+16)

    v8f acc[4][4];
    #pragma unroll
    for (int i = 0; i < 4; i++)
        #pragma unroll
        for (int j = 0; j < 4; j++) acc[i][j] = (v8f){0,0,0,0,0,0,0,0};

    for (int k0 = 0; k0 < DIM; k0 += 32) {
        v16bf a[4], b[4];
        #pragma unroll
        for (int i = 0; i < 4; i++)
            a[i] = load_frag(xb + arow[i] + k0 + aoff,
                             xb + arow[i] + k0 + aoff + 16);
        #pragma unroll
        for (int j = 0; j < 4; j++)
            b[j] = load_frag(wbase + brow[j] + k0 + boff,
                             wbase + brow[j] + k0 + boff + 16);
        #pragma unroll
        for (int i = 0; i < 4; i++)
            #pragma unroll
            for (int j = 0; j < 4; j++)
                acc[i][j] = __builtin_amdgcn_wmma_f32_16x16x32_bf16(
                    false, a[i], false, b[j], (short)0, acc[i][j], false, false);
    }

    const long hb = (long)basep[e];
    #pragma unroll
    for (int j = 0; j < 4; j++) {
        int n = nW + j*16 + laneLo;
        float bias = b1[(long)e * HID + n];
        #pragma unroll
        for (int i = 0; i < 4; i++) {
            #pragma unroll
            for (int r = 0; r < 8; r++) {
                int s = mW + i*16 + laneHi*8 + r;   // C layout: M = hi*8 + r
                if (s < cnt) {
                    float v = acc[i][j][r] + bias;
                    v = v / (1.f + __expf(-v));     // Swish
                    h[(hb + s) * (long)HID + n] = f2bf(v);
                }
            }
        }
    }
}

// ---------------------------------------------------------------------------
// K8: expert GEMM2 + bias: y = H @ W2^T + b2, fp32 out. Same tiling, K = HID.
// A rows are the compacted h rows (contiguous, no gather needed).
// ---------------------------------------------------------------------------
__global__ __launch_bounds__(256)
void k_gemm2_bias(const unsigned short* __restrict__ h,
                  const unsigned short* __restrict__ w2b,
                  const float* __restrict__ b2,
                  const int* __restrict__ counts,
                  const int* __restrict__ basep,
                  float* __restrict__ y) {
    const int e    = blockIdx.z;
    const int cnt  = counts[e];
    const int mblk = blockIdx.y * 128;
    if (mblk >= cnt) return;
    const int nblk = blockIdx.x * 256;
    const int wave = threadIdx.x >> 5;
    const int lane = threadIdx.x & 31;
    const int laneLo = lane & 15, laneHi = lane >> 4;
    const int wm = wave >> 2, wn = wave & 3;
    const int mW = mblk + wm * 64;
    const int nW = nblk + wn * 64;
    const long hb = (long)basep[e];

    long arow[4];
    #pragma unroll
    for (int i = 0; i < 4; i++) {
        int s = mW + i*16 + laneLo; if (s >= cnt) s = cnt - 1;
        arow[i] = (hb + s) * (long)HID;
    }
    const unsigned short* wbase = w2b + (long)e * DIM * HID;
    long brow[4];
    #pragma unroll
    for (int j = 0; j < 4; j++) brow[j] = (long)(nW + j*16 + laneLo) * HID;

    const int aoff = laneHi ? 8 : 0;
    const int boff = laneHi ? 8 : 0;

    v8f acc[4][4];
    #pragma unroll
    for (int i = 0; i < 4; i++)
        #pragma unroll
        for (int j = 0; j < 4; j++) acc[i][j] = (v8f){0,0,0,0,0,0,0,0};

    for (int k0 = 0; k0 < HID; k0 += 32) {
        v16bf a[4], b[4];
        #pragma unroll
        for (int i = 0; i < 4; i++)
            a[i] = load_frag(h + arow[i] + k0 + aoff,
                             h + arow[i] + k0 + aoff + 16);
        #pragma unroll
        for (int j = 0; j < 4; j++)
            b[j] = load_frag(wbase + brow[j] + k0 + boff,
                             wbase + brow[j] + k0 + boff + 16);
        #pragma unroll
        for (int i = 0; i < 4; i++)
            #pragma unroll
            for (int j = 0; j < 4; j++)
                acc[i][j] = __builtin_amdgcn_wmma_f32_16x16x32_bf16(
                    false, a[i], false, b[j], (short)0, acc[i][j], false, false);
    }

    #pragma unroll
    for (int j = 0; j < 4; j++) {
        int n = nW + j*16 + laneLo;
        float bias = b2[(long)e * DIM + n];
        #pragma unroll
        for (int i = 0; i < 4; i++) {
            #pragma unroll
            for (int r = 0; r < 8; r++) {
                int s = mW + i*16 + laneHi*8 + r;
                if (s < cnt)
                    y[(hb + s) * (long)DIM + n] = acc[i][j][r] + bias;
            }
        }
    }
}

// ---------------------------------------------------------------------------
// K9: combine the two expert outputs per token (each token has exactly 2
// experts -> no atomics). One wave32 per token row.
// ---------------------------------------------------------------------------
__global__ void k_combine(const float* __restrict__ y,
                          const int* __restrict__ tok_map,
                          const int* __restrict__ basep,
                          float* __restrict__ out) {
    int wave = (int)((blockIdx.x * blockDim.x + threadIdx.x) >> 5);
    int lane = threadIdx.x & 31;
    if (wave >= T_TOK) return;
    int e0 = tok_map[wave*4 + 0], s0 = tok_map[wave*4 + 1];
    int e1 = tok_map[wave*4 + 2], s1 = tok_map[wave*4 + 3];
    const float4* y0 = (const float4*)(y + (long)(basep[e0] + s0) * DIM);
    const float4* y1 = (const float4*)(y + (long)(basep[e1] + s1) * DIM);
    float4* o = (float4*)(out + (long)wave * DIM);
    #pragma unroll 2
    for (int i = lane; i < DIM/4; i += 32) {
        float4 a = y0[i], b = y1[i], c;
        c.x = a.x + b.x; c.y = a.y + b.y; c.z = a.z + b.z; c.w = a.w + b.w;
        o[i] = c;
    }
}

// ---------------------------------------------------------------------------
// host launcher
// ---------------------------------------------------------------------------
extern "C" void kernel_launch(void* const* d_in, const int* in_sizes, int n_in,
                              void* d_out, int out_size, void* d_ws, size_t ws_size,
                              hipStream_t stream) {
    (void)in_sizes; (void)n_in; (void)out_size; (void)ws_size;
    const float* x      = (const float*)d_in[0];
    const float* gate_v = (const float*)d_in[1];
    const float* gate_g = (const float*)d_in[2];
    const float* gate_b = (const float*)d_in[3];
    const float* w1_v   = (const float*)d_in[4];
    const float* w1_g   = (const float*)d_in[5];
    const float* b1     = (const float*)d_in[6];
    const float* w2_v   = (const float*)d_in[7];
    const float* w2_g   = (const float*)d_in[8];
    const float* b2     = (const float*)d_in[9];
    float* out = (float*)d_out;

    // workspace carve-up (~337 MB total)
    char* ws = (char*)d_ws;
    size_t off = 0;
    auto take = [&](size_t bytes) -> char* {
        char* p = ws + off;
        off = (off + bytes + 255) & ~(size_t)255;
        return p;
    };
    int*            counts  = (int*)take(NE * 4);
    int*            basep   = (int*)take(NE * 4);
    int*            tok_idx = (int*)take((size_t)NE * T_TOK * 4);
    int*            tok_map = (int*)take((size_t)T_TOK * 16);
    float*          gwn     = (float*)take((size_t)NE * DIM * 4);
    unsigned short* xb      = (unsigned short*)take((size_t)T_TOK * DIM * 2);
    unsigned short* w1b     = (unsigned short*)take((size_t)NE * HID * DIM * 2);
    unsigned short* w2b     = (unsigned short*)take((size_t)NE * DIM * HID * 2);
    unsigned short* hbuf    = (unsigned short*)take((size_t)2 * T_TOK * HID * 2);
    float*          ybuf    = (float*)take((size_t)2 * T_TOK * DIM * 4);

    k_zero_counts<<<1, 32, 0, stream>>>(counts);
    k_rownorm_f32 <<<1, 256, 0, stream>>>(gate_v, gate_g, gwn, NE, DIM);
    k_rownorm_bf16<<<(NE * HID) / 8, 256, 0, stream>>>(w1_v, w1_g, w1b, NE * HID, DIM);
    k_rownorm_bf16<<<(NE * DIM) / 8, 256, 0, stream>>>(w2_v, w2_g, w2b, NE * DIM, HID);
    k_cvt_x      <<<T_TOK / 8, 256, 0, stream>>>(x, xb);
    k_gate_route <<<T_TOK / 8, 256, 0, stream>>>(x, gwn, gate_b, counts, tok_idx, tok_map);
    k_prefix     <<<1, 32, 0, stream>>>(counts, basep);

    dim3 g1(HID / 256, T_TOK / 128, NE);   // (16, 64, 8), most M-tiles early-exit
    k_gemm1_swish<<<g1, 256, 0, stream>>>(xb, w1b, b1, counts, basep, tok_idx, hbuf);

    dim3 g2(DIM / 256, T_TOK / 128, NE);   // (4, 64, 8)
    k_gemm2_bias <<<g2, 256, 0, stream>>>(hbuf, w2b, b2, counts, basep, ybuf);

    k_combine    <<<T_TOK / 8, 256, 0, stream>>>(ybuf, tok_map, basep, out);
}
